// MetricLoss_20134806684259
// MI455X (gfx1250) — matrix operand order repaired and checked by reading, not compiled, not run on hardware
//
#include <hip/hip_runtime.h>

// ---------------------------------------------------------------------------
// Problem constants (fixed by the reference: x is [1024, 64, 768] fp32)
// ---------------------------------------------------------------------------
#define NS    1024        // number of samples (rows of F)
#define KD    49152       // flattened feature dim (64*768)
#define TILE  64          // output tile edge per workgroup
#define KC    64          // K-chunk staged in LDS per iteration
#define LSTR  72          // LDS row stride in bf16 elems (64 + 8 pad, 16B-aligned rows)
#define BK    8           // samples per class group
#define NGRP  (NS / BK)   // 128 groups

typedef __attribute__((ext_vector_type(16))) __bf16 v16bf;
typedef __attribute__((ext_vector_type(8)))  __bf16 v8bf;
typedef __attribute__((ext_vector_type(4)))  __bf16 v4bf;
typedef __attribute__((ext_vector_type(2)))  __bf16 v2bf;
typedef __attribute__((ext_vector_type(8)))  float  v8f;
typedef __attribute__((ext_vector_type(4)))  float  v4f;

// ---------------------------------------------------------------------------
// fp32x4 -> bf16x4 (RNE).  A vector fptrunc lets the backend select the
// packed hardware converter (v_cvt_pk_bf16_f32) when the target has it;
// otherwise clang emits the generic RNE sequence (same cost as the manual
// bit-trick).  If the named builtin exists, use it directly.
// ---------------------------------------------------------------------------
#if __has_builtin(__builtin_amdgcn_cvt_pk_bf16_f32)
#define HAVE_HW_PK_BF16 1
#endif

__device__ __forceinline__ v4bf pack4_bf16(v4f v) {
#ifdef HAVE_HW_PK_BF16
    v2bf lo = __builtin_amdgcn_cvt_pk_bf16_f32(v.x, v.y);
    v2bf hi = __builtin_amdgcn_cvt_pk_bf16_f32(v.z, v.w);
    return __builtin_shufflevector(lo, hi, 0, 1, 2, 3);
#else
    return __builtin_convertvector(v, v4bf);   // vector fptrunc, RNE
#endif
}

// ---------------------------------------------------------------------------
// Kernel 1: G = F * F^T via v_wmma_f32_16x16x32_bf16
//   grid  = 256 (16x16 tile grid), block = 128 (4 waves)
//   wave w computes rows [i0 + 16w, i0 + 16w + 15] x cols [j0, j0+63]
// ---------------------------------------------------------------------------
__global__ void __launch_bounds__(128)
gram_bf16_wmma(const float* __restrict__ x, float* __restrict__ G)
{
    __shared__ __bf16 As[2][TILE * LSTR];   // 9 KB  * 2 buffers
    __shared__ __bf16 Bs[2][TILE * LSTR];   // 9 KB  * 2 buffers  (36 KB total)

    const int tid  = threadIdx.x;
    const int ti   = blockIdx.x >> 4;
    const int tj   = blockIdx.x & 15;
    const int i0   = ti * TILE;
    const int j0   = tj * TILE;
    const int wave = tid >> 5;
    const int lane = tid & 31;
    const int m    = lane & 15;       // row (A) / col (B,D) within 16
    const int hp   = lane >> 4;       // half-wave selector for K / M split

    // ---- per-thread global staging: 8 float4 per panel (64 rows x 64 k / 128 thr)
    v4f sa[8], sb[8];

    auto gload = [&](int kc) {
        const float* __restrict__ ab = x + (size_t)i0 * KD + kc;
        const float* __restrict__ bb = x + (size_t)j0 * KD + kc;
#pragma unroll
        for (int u = 0; u < 8; ++u) {
            int idx = tid + 128 * u;            // 0..1023
            int row = idx >> 4;
            int c4  = (idx & 15) << 2;          // float offset in chunk
            sa[u] = *(const v4f*)(ab + (size_t)row * KD + c4);
            sb[u] = *(const v4f*)(bb + (size_t)row * KD + c4);
        }
    };
    auto lstore = [&](int buf) {
#pragma unroll
        for (int u = 0; u < 8; ++u) {
            int idx = tid + 128 * u;
            int row = idx >> 4;
            int c4  = (idx & 15) << 2;
            *(v4bf*)&As[buf][row * LSTR + c4] = pack4_bf16(sa[u]);
            *(v4bf*)&Bs[buf][row * LSTR + c4] = pack4_bf16(sb[u]);
        }
    };

    v8f acc[4] = {};

    // prologue: stage chunk 0
    gload(0);
    lstore(0);

    for (int kc = 0; kc < KD; kc += KC) {
        const int buf = (kc / KC) & 1;

        // issue next chunk's global loads early (L2 latency overlaps the WMMAs)
        if (kc + KC < KD) gload(kc + KC);

        __syncthreads();                        // chunk kc now resident in LDS[buf]

        const __bf16* a_base = &As[buf][(wave * 16 + m) * LSTR];
        const __bf16* b_base = &Bs[buf][0];

#pragma unroll
        for (int ks = 0; ks < KC; ks += 32) {
            // A fragment (16x32 bf16, contiguous K, same order as B):
            //   lane<16 : K = ks+0..15
            //   lane>=16: K = ks+16..31
            v8bf a_lo = *(const v8bf*)(a_base + ks + hp * 16);
            v8bf a_hi = *(const v8bf*)(a_base + ks + hp * 16 + 8);
            v16bf a = __builtin_shufflevector(a_lo, a_hi,
                        0,1,2,3,4,5,6,7,8,9,10,11,12,13,14,15);
#pragma unroll
            for (int t = 0; t < 4; ++t) {
                // B fragment (32x16): lane holds 16 contiguous K of column (t*16+m),
                // halves split at K=16.  B[k][n] = F[j0 + n][k].
                const __bf16* bp = b_base + (t * 16 + m) * LSTR + ks + hp * 16;
                v8bf b_lo = *(const v8bf*)(bp);
                v8bf b_hi = *(const v8bf*)(bp + 8);
                v16bf b = __builtin_shufflevector(b_lo, b_hi,
                            0,1,2,3,4,5,6,7,8,9,10,11,12,13,14,15);
                acc[t] = __builtin_amdgcn_wmma_f32_16x16x32_bf16(
                            false, a, false, b, (short)0, acc[t], false, false);
            }
        }

        // convert + stage next chunk into the other LDS buffer (DS is in-order,
        // so this never delays the fragment loads above)
        if (kc + KC < KD) lstore(buf ^ 1);
    }

    // D layout: vgpr r, lanes<16 -> M=r, lanes>=16 -> M=r+8; N = m
    const int row0 = i0 + wave * 16 + hp * 8;
#pragma unroll
    for (int t = 0; t < 4; ++t) {
        const int col = j0 + t * 16 + m;
#pragma unroll
        for (int r = 0; r < 8; ++r) {
            G[(size_t)(row0 + r) * NS + col] = acc[t][r];
        }
    }
}

// ---------------------------------------------------------------------------
// Kernel 2: per-block partial loss sums over the 1024x1024 pair matrix.
//   grid = 256, block = 256; each thread handles 16 entries (coalesced in j).
//   dist_ij = max(G_ii + G_jj - 2*G_ij, 0)   (diag of G supplies the norms)
// ---------------------------------------------------------------------------
__global__ void __launch_bounds__(256)
loss_partial(const float* __restrict__ G, float* __restrict__ part)
{
    __shared__ float diag[NS];      // 4 KB: cached G diagonal
    __shared__ float shh[256];
    __shared__ float she[256];

    const int tid = threadIdx.x;
#pragma unroll
    for (int u = 0; u < NS / 256; ++u) {
        int i = tid + u * 256;
        diag[i] = G[(size_t)i * NS + i];
    }
    __syncthreads();

    const int g = blockIdx.x * 256 + tid;   // 0..65535
    float h = 0.0f, e = 0.0f;
#pragma unroll
    for (int it = 0; it < 16; ++it) {
        int idx = g + 65536 * it;           // covers all 1M entries exactly once
        int i = idx >> 10;
        int j = idx & (NS - 1);
        float gij  = G[idx];
        float dist = fmaxf(diag[i] + diag[j] - 2.0f * gij, 0.0f);
        if ((i >> 3) == (j >> 3)) {         // same group
            if (i != j) h += dist;
        } else {                            // different group: hinge
            e += fmaxf(1.0f - dist, 0.0f);
        }
    }

    shh[tid] = h; she[tid] = e;
    __syncthreads();
#pragma unroll
    for (int s = 128; s > 0; s >>= 1) {
        if (tid < s) { shh[tid] += shh[tid + s]; she[tid] += she[tid + s]; }
        __syncthreads();
    }
    if (tid == 0) { part[blockIdx.x] = shh[0]; part[256 + blockIdx.x] = she[0]; }
}

// ---------------------------------------------------------------------------
// Kernel 3: reduce 256 partials, apply reference scalings, write 2 outputs.
//   out0 = (0.5*S_same)*2/(N*(BK-1))   = S_same / (N*7)
//   out1 = 2*S_heter/(N*(NGRP-1))      = 2*S_heter / (N*127)
// ---------------------------------------------------------------------------
__global__ void __launch_bounds__(256)
loss_finalize(const float* __restrict__ part, float* __restrict__ out)
{
    __shared__ float shh[256];
    __shared__ float she[256];
    const int tid = threadIdx.x;
    shh[tid] = part[tid];
    she[tid] = part[256 + tid];
    __syncthreads();
#pragma unroll
    for (int s = 128; s > 0; s >>= 1) {
        if (tid < s) { shh[tid] += shh[tid + s]; she[tid] += she[tid + s]; }
        __syncthreads();
    }
    if (tid == 0) {
        out[0] = shh[0] * (1.0f / ((float)NS * (float)(BK - 1)));
        out[1] = she[0] * (2.0f / ((float)NS * (float)(NGRP - 1)));
    }
}

// ---------------------------------------------------------------------------
// Host launcher
// ---------------------------------------------------------------------------
extern "C" void kernel_launch(void* const* d_in, const int* in_sizes, int n_in,
                              void* d_out, int out_size, void* d_ws, size_t ws_size,
                              hipStream_t stream)
{
    (void)in_sizes; (void)n_in; (void)out_size; (void)ws_size;

    const float* x = (const float*)d_in[0];        // [1024, 64, 768] fp32
    float* G    = (float*)d_ws;                    // 1024*1024 fp32 = 4 MB
    float* part = (float*)((char*)d_ws + (size_t)NS * NS * sizeof(float)); // 2*256 fp32

    gram_bf16_wmma<<<dim3(256), dim3(128), 0, stream>>>(x, G);
    loss_partial  <<<dim3(256), dim3(256), 0, stream>>>(G, part);
    loss_finalize <<<dim3(1),   dim3(256), 0, stream>>>(part, (float*)d_out);
}
